// SparseLinearLayer_1700807049625
// MI455X (gfx1250) — matrix operand order repaired and checked by reading, not compiled, not run on hardware
//
#include <hip/hip_runtime.h>
#include <hip/hip_bf16.h>

#define B_ROWS 8192
#define D_IN   64
#define D_OUT  32000

typedef __attribute__((ext_vector_type(16))) __bf16 v16bf;
typedef __attribute__((ext_vector_type(8)))  __bf16 v8bf;
typedef __attribute__((ext_vector_type(8)))  float  v8f;

union ABf { v16bf v; v8bf h[2]; };

#define WMMA_BF16(A, B, C) \
  __builtin_amdgcn_wmma_f32_16x16x32_bf16(false, (A), false, (B), (short)0, (C), false, false)

__device__ __forceinline__ unsigned short f32_to_bf16_rne(float f) {
  unsigned int u = __float_as_uint(f);
  u += 0x7FFFu + ((u >> 16) & 1u);   // round-to-nearest-even
  return (unsigned short)(u >> 16);
}

// Convert 1/x and W to bf16 in workspace, zero row sums, write aux_loss (== 0).
__global__ void prep_kernel(const float* __restrict__ x,
                            const float* __restrict__ w,
                            unsigned short* __restrict__ rx_bf,
                            unsigned short* __restrict__ w_bf,
                            float* __restrict__ rowsum,
                            float* __restrict__ aux_out) {
  const int NRX = B_ROWS * D_IN;   // 524288
  const int NW  = D_IN * D_OUT;    // 2048000
  int i = blockIdx.x * blockDim.x + threadIdx.x;
  if (i < NRX) {
    rx_bf[i] = f32_to_bf16_rne(1.0f / x[i]);
  } else if (i < NRX + NW) {
    w_bf[i - NRX] = f32_to_bf16_rne(w[i - NRX]);
  } else if (i < NRX + NW + B_ROWS) {
    rowsum[i - NRX - NW] = 0.0f;
  }
  if (i == 0) *aux_out = 0.0f;     // aux_loss is identically zero
}

// Load the two 16x32 bf16 A fragments (K 0..31, 32..63) for one 16-row tile.
__device__ __forceinline__ void load_a_frags(const __bf16* rx, int row0,
                                             int m, int khalf, ABf& a0, ABf& a1) {
  const __bf16* ap = rx + (size_t)(row0 + m) * D_IN;
  a0.h[0] = *reinterpret_cast<const v8bf*>(ap +      khalf * 8);   // K 0..7 | 8..15
  a0.h[1] = *reinterpret_cast<const v8bf*>(ap + 16 + khalf * 8);   // K 16..23 | 24..31
  a1.h[0] = *reinterpret_cast<const v8bf*>(ap + 32 + khalf * 8);
  a1.h[1] = *reinterpret_cast<const v8bf*>(ap + 48 + khalf * 8);
}

// Pass 1: per-row sum of exp(product) via online WMMA recompute. Each wave owns
// a 32x128 output tile (two accumulator tiles share each B fragment -> half the
// L2 read traffic on W). Only 32 KB of f32 atomics reach memory.
// |product| <= 64 * 2 * 0.05 = 6.4, so the max-shift is unnecessary (exact softmax).
__global__ __launch_bounds__(32) void gemm_expsum_kernel(
    const unsigned short* __restrict__ rx_bf,
    const unsigned short* __restrict__ w_bf,
    float* __restrict__ rowsum) {
  const int lane  = threadIdx.x;
  const int row0  = blockIdx.y << 5;   // 32 rows / block (2 row tiles)
  const int col0  = blockIdx.x << 7;   // 128 cols / block
  const int m     = lane & 15;
  const int khalf = lane >> 4;

  const __bf16* rx = reinterpret_cast<const __bf16*>(rx_bf);
  ABf a0_0, a1_0, a0_1, a1_1;
  load_a_frags(rx, row0,      m, khalf, a0_0, a1_0);
  load_a_frags(rx, row0 + 16, m, khalf, a0_1, a1_1);

  const __bf16* wp0 = reinterpret_cast<const __bf16*>(w_bf) + (size_t)(khalf * 8) * D_OUT + m; // K base = 8*khalf, N = m
  const __bf16* wp1 = wp0 + (size_t)32 * D_OUT;                                     // K base + 32

  float s0[8], s1[8];
#pragma unroll
  for (int v = 0; v < 8; ++v) { s0[v] = 0.0f; s1[v] = 0.0f; }

#pragma unroll
  for (int t = 0; t < 8; ++t) {
    const int n0 = col0 + t * 16;
    ABf b0, b1;
#pragma unroll
    for (int s = 0; s < 16; ++s) {
      const size_t ko = (size_t)(s + (s & 8)) * D_OUT + n0;
      b0.v[s] = wp0[ko];
      b1.v[s] = wp1[ko];
    }
    v8f c0 = {}, c1 = {};
    c0 = WMMA_BF16(a0_0.v, b0.v, c0);
    c1 = WMMA_BF16(a0_1.v, b0.v, c1);
    c0 = WMMA_BF16(a1_0.v, b1.v, c0);
    c1 = WMMA_BF16(a1_1.v, b1.v, c1);
#pragma unroll
    for (int v = 0; v < 8; ++v) { s0[v] += __expf(c0[v]); s1[v] += __expf(c1[v]); }
  }

  // C layout: vgpr v holds row (khalf*8+v), N = lane&15 -> reduce across 16 lanes.
#pragma unroll
  for (int v = 0; v < 8; ++v) {
    float t0 = s0[v], t1 = s1[v];
    t0 += __shfl_xor(t0, 1, 32);  t1 += __shfl_xor(t1, 1, 32);
    t0 += __shfl_xor(t0, 2, 32);  t1 += __shfl_xor(t1, 2, 32);
    t0 += __shfl_xor(t0, 4, 32);  t1 += __shfl_xor(t1, 4, 32);
    t0 += __shfl_xor(t0, 8, 32);  t1 += __shfl_xor(t1, 8, 32);
    s0[v] = t0; s1[v] = t1;
  }
  if (m == 0) {
    const int rbase = row0 + khalf * 8;
#pragma unroll
    for (int v = 0; v < 8; ++v) {
      atomicAdd(&rowsum[rbase + v],      s0[v]);
      atomicAdd(&rowsum[rbase + 16 + v], s1[v]);
    }
  }
}

// Pass 2: recompute product (bf16 operands are L2-resident, 5 MB total) and
// stream exp(p)/rowsum to HBM with NON-TEMPORAL stores so the 1.05 GB output
// (write-once, never re-read) does not evict the operands from L2.
// HBM floor: 1.05 GB / 23.3 TB/s ~= 45 us.
__global__ __launch_bounds__(32) void gemm_softmax_kernel(
    const unsigned short* __restrict__ rx_bf,
    const unsigned short* __restrict__ w_bf,
    const float* __restrict__ rowsum,
    float* __restrict__ out) {
  const int lane  = threadIdx.x;
  const int row0  = blockIdx.y << 5;
  const int col0  = blockIdx.x << 7;
  const int m     = lane & 15;
  const int khalf = lane >> 4;

  const __bf16* rx = reinterpret_cast<const __bf16*>(rx_bf);
  ABf a0_0, a1_0, a0_1, a1_1;
  load_a_frags(rx, row0,      m, khalf, a0_0, a1_0);
  load_a_frags(rx, row0 + 16, m, khalf, a0_1, a1_1);

  const __bf16* wp0 = reinterpret_cast<const __bf16*>(w_bf) + (size_t)(khalf * 8) * D_OUT + m;
  const __bf16* wp1 = wp0 + (size_t)32 * D_OUT;

  const int rbase = row0 + khalf * 8;
  float rinv0[8], rinv1[8];
#pragma unroll
  for (int v = 0; v < 8; ++v) {
    rinv0[v] = 1.0f / rowsum[rbase + v];
    rinv1[v] = 1.0f / rowsum[rbase + 16 + v];
  }

#pragma unroll
  for (int t = 0; t < 8; ++t) {
    const int n0 = col0 + t * 16;
    ABf b0, b1;
#pragma unroll
    for (int s = 0; s < 16; ++s) {
      const size_t ko = (size_t)(s + (s & 8)) * D_OUT + n0;
      b0.v[s] = wp0[ko];
      b1.v[s] = wp1[ko];
    }
    v8f c0 = {}, c1 = {};
    c0 = WMMA_BF16(a0_0.v, b0.v, c0);
    c1 = WMMA_BF16(a0_1.v, b0.v, c1);
    c0 = WMMA_BF16(a1_0.v, b1.v, c0);
    c1 = WMMA_BF16(a1_1.v, b1.v, c1);
    const int col = n0 + m;
#pragma unroll
    for (int v = 0; v < 8; ++v) {
      __builtin_nontemporal_store(__expf(c0[v]) * rinv0[v],
                                  out + (size_t)(rbase + v) * D_OUT + col);
      __builtin_nontemporal_store(__expf(c1[v]) * rinv1[v],
                                  out + (size_t)(rbase + 16 + v) * D_OUT + col);
    }
  }
}

extern "C" void kernel_launch(void* const* d_in, const int* in_sizes, int n_in,
                              void* d_out, int out_size, void* d_ws, size_t ws_size,
                              hipStream_t stream) {
  const float* x = (const float*)d_in[0];   // (8192, 64)
  const float* w = (const float*)d_in[1];   // (64, 32000) ; sparsity_weights (d_in[2]) unused
  float* out = (float*)d_out;               // 8192*32000 softmax + 1 aux scalar

  // Workspace layout (all 256B-aligned): rx_bf 1MB | w_bf 4,096,000B | rowsum 32KB
  unsigned char* ws = (unsigned char*)d_ws;
  unsigned short* rx_bf  = (unsigned short*)(ws);
  unsigned short* w_bf   = (unsigned short*)(ws + (1u << 20));
  float*          rowsum = (float*)(ws + (1u << 20) + 4096000u);   // offset 5,144,576 (%256==0)

  float* aux = out + (size_t)B_ROWS * D_OUT;

  const int NPREP = B_ROWS * D_IN + D_IN * D_OUT + B_ROWS;
  prep_kernel<<<(NPREP + 255) / 256, 256, 0, stream>>>(x, w, rx_bf, w_bf, rowsum, aux);

  dim3 grid(D_OUT / 128, B_ROWS / 32);   // (250, 256), one wave32 per block, 32x128 tile
  gemm_expsum_kernel<<<grid, 32, 0, stream>>>(rx_bf, w_bf, rowsum);
  gemm_softmax_kernel<<<grid, 32, 0, stream>>>(rx_bf, w_bf, rowsum, out);
}
